// RNN_53824530153961
// MI455X (gfx1250) — compile-verified, matrix operand order checked by reading, not run on hardware
//
#include <hip/hip_runtime.h>
#include <math.h>

// Problem constants (match reference)
#define T_STEPS 128
#define BATCH   64
#define EMB     1024
#define HID     1024
#define VOC     10000
#define VOCP    10240              // vocab padded to multiple of 64 for tiling
#define MROWS   (T_STEPS * BATCH)  // 8192

// ---------------------------------------------------------------------------
// WMMA types / helpers (CDNA5 v_wmma_f32_16x16x32_bf16, wave32)
// ---------------------------------------------------------------------------
typedef __attribute__((ext_vector_type(16))) __bf16 bf16x16;
typedef __attribute__((ext_vector_type(8)))  float  f32x8;

union FragU { bf16x16 v; uint4 u[2]; };

__device__ __forceinline__ unsigned short f32_to_bf16(float f) {
  unsigned u = __float_as_uint(f);
  u += 0x7FFFu + ((u >> 16) & 1u);   // round-to-nearest-even
  return (unsigned short)(u >> 16);
}

// Load a 16-element bf16 fragment for one lane from a row-major row.
// k already includes the per-half-wave +8 shift: lanes 0-15 carry
// K {k..k+7, k+16..k+23}, lanes 16-31 the other half.
__device__ __forceinline__ bf16x16 load_frag(const unsigned short* __restrict rowp, int k) {
  FragU f;
  f.u[0] = *(const uint4*)(rowp + k);        // 8 bf16
  f.u[1] = *(const uint4*)(rowp + k + 16);   // 8 bf16
  return f.v;
}

__device__ __forceinline__ f32x8 wmma_bf16(bf16x16 a, bf16x16 b, f32x8 c) {
  return __builtin_amdgcn_wmma_f32_16x16x32_bf16(false, a, false, b, (short)0, c,
                                                 false, false);
}

// ---------------------------------------------------------------------------
// Conversion / init kernels
// ---------------------------------------------------------------------------
__global__ void cvt_f32_bf16_kernel(const float* __restrict src,
                                    unsigned short* __restrict dst, int n) {
  for (int i = blockIdx.x * blockDim.x + threadIdx.x; i < n;
       i += gridDim.x * blockDim.x)
    dst[i] = f32_to_bf16(src[i]);
}

// Wdec [VOC,HID] -> bf16 padded to [VOCP,HID] (pad rows zero)
__global__ void cvt_wdec_pad_kernel(const float* __restrict src,
                                    unsigned short* __restrict dst) {
  const int total = VOCP * HID;
  for (int i = blockIdx.x * blockDim.x + threadIdx.x; i < total;
       i += gridDim.x * blockDim.x) {
    int row = i >> 10;
    dst[i] = (row < VOC) ? f32_to_bf16(src[i]) : (unsigned short)0;
  }
}

struct GridBar { unsigned count; unsigned gen; };

// hidden [L=2, B, H] f32 -> hstate0 [B, 2048] bf16 (h0 cols 0..1023, h1 cols 1024..)
__global__ void init_state_kernel(const float* __restrict hidden,
                                  unsigned short* __restrict hst0,
                                  unsigned short* __restrict hst1,
                                  GridBar* bar) {
  const int tid = blockIdx.x * blockDim.x + threadIdx.x;
  if (tid == 0) { bar->count = 0u; bar->gen = 0u; }
  const int total = BATCH * 2 * HID;
  for (int i = tid; i < total; i += gridDim.x * blockDim.x) {
    int b = i >> 11;        // / 2048
    int c = i & 2047;
    int l = c >> 10;
    int h = c & 1023;
    hst0[i] = f32_to_bf16(hidden[(size_t)l * BATCH * HID + (size_t)b * HID + h]);
    hst1[i] = 0;
  }
}

// ---------------------------------------------------------------------------
// GEMM 1: A0[t*B+b, n] = b0[n] + sum_k emb[tokens[t,b],k] * W_in0[n,k]
// M=8192 N=1024 K=1024, bf16 WMMA, f32 out. Register-direct (only ~10% of
// decoder FLOPs; A rows are token-gathered so keep it simple).
// ---------------------------------------------------------------------------
__global__ void __launch_bounds__(256) gemm_a0_kernel(
    const unsigned short* __restrict embb,  // [VOC, EMB] bf16
    const unsigned short* __restrict w0b,   // [HID, EMB] bf16 (N,K)
    const float* __restrict b0,
    const int* __restrict tokens,           // [T*B]
    float* __restrict A0) {                 // [MROWS, HID]
  const int lane = threadIdx.x & 31;
  const int wave = threadIdx.x >> 5;
  const int hsel = lane >> 4;
  const int l15  = lane & 15;
  const int m0 = blockIdx.x * 64 + (wave & 3) * 16;
  const int n0 = blockIdx.y * 64 + (wave >> 2) * 32;

  const unsigned short* arow = embb + (size_t)tokens[m0 + l15] * EMB;
  const unsigned short* br0  = w0b + (size_t)(n0 + l15) * EMB;
  const unsigned short* br1  = br0 + (size_t)16 * EMB;

  f32x8 acc0 = {}; f32x8 acc1 = {};
  for (int k = 0; k < EMB; k += 32) {
    __builtin_prefetch(br0 + k + 512, 0, 1);  // global_prefetch ahead in K
    bf16x16 a = load_frag(arow, k + hsel * 8);
    acc0 = wmma_bf16(a, load_frag(br0, k + hsel * 8), acc0);
    acc1 = wmma_bf16(a, load_frag(br1, k + hsel * 8), acc1);
  }
  const int n_a = n0 + l15, n_b = n_a + 16;
  const float ba = b0[n_a], bb = b0[n_b];
#pragma unroll
  for (int j = 0; j < 8; ++j) {
    int m = m0 + hsel * 8 + j;   // C layout: VGPR j -> M = j (+8 for hi half)
    A0[(size_t)m * HID + n_a] = acc0[j] + ba;
    A0[(size_t)m * HID + n_b] = acc1[j] + bb;
  }
}

// ---------------------------------------------------------------------------
// Persistent recurrent kernel: 16 resident blocks, device-wide sense barrier.
// Block bk owns output columns [bk*64, bk*64+64) of each layer.
// ---------------------------------------------------------------------------
__device__ __forceinline__ void grid_barrier(GridBar* bar, unsigned nblocks) {
  __threadfence();
  __syncthreads();
  if (threadIdx.x == 0) {
    unsigned my = __hip_atomic_load(&bar->gen, __ATOMIC_RELAXED,
                                    __HIP_MEMORY_SCOPE_AGENT);
    unsigned prev = __hip_atomic_fetch_add(&bar->count, 1u, __ATOMIC_ACQ_REL,
                                           __HIP_MEMORY_SCOPE_AGENT);
    if (prev == nblocks - 1u) {
      __hip_atomic_store(&bar->count, 0u, __ATOMIC_RELAXED,
                         __HIP_MEMORY_SCOPE_AGENT);
      __hip_atomic_fetch_add(&bar->gen, 1u, __ATOMIC_RELEASE,
                             __HIP_MEMORY_SCOPE_AGENT);
    } else {
      while (__hip_atomic_load(&bar->gen, __ATOMIC_ACQUIRE,
                               __HIP_MEMORY_SCOPE_AGENT) == my)
        __builtin_amdgcn_s_sleep(1);
    }
  }
  __syncthreads();
}

__global__ void __launch_bounds__(256) rnn_recurrent_kernel(
    const float* __restrict A0,            // [MROWS, HID] (x@W_in0^T + b0)
    const unsigned short* __restrict wh0b, // [HID, HID] bf16
    const unsigned short* __restrict w1b,  // [HID, HID] bf16 (W_in1)
    const unsigned short* __restrict wh1b, // [HID, HID] bf16
    const float* __restrict b1,
    unsigned short* hstA, unsigned short* hstB,  // [BATCH, 2048] bf16 ping-pong
    unsigned short* __restrict H1all,            // [MROWS, HID] bf16
    float* __restrict hfinal,                    // [2, BATCH, HID] f32
    GridBar* bar) {
  const int lane = threadIdx.x & 31;
  const int wave = threadIdx.x >> 5;
  const int hsel = lane >> 4;
  const int l15  = lane & 15;
  const int m0 = (wave & 3) * 16;
  const int n0 = blockIdx.x * 64 + (wave >> 2) * 32;
  const int n_a = n0 + l15, n_b = n_a + 16;
  const int arow_m = m0 + l15;

  const unsigned short* wr0a = wh0b + (size_t)n_a * HID;
  const unsigned short* wr0b = wh0b + (size_t)n_b * HID;
  const unsigned short* wr1a = w1b  + (size_t)n_a * HID;
  const unsigned short* wr1b = w1b  + (size_t)n_b * HID;
  const unsigned short* wr2a = wh1b + (size_t)n_a * HID;
  const unsigned short* wr2b = wh1b + (size_t)n_b * HID;
  const float b1a = b1[n_a], b1b = b1[n_b];

  unsigned short* cur = hstA;
  unsigned short* nxt = hstB;

  for (int t = 0; t < T_STEPS; ++t) {
    // ---- layer 0: h0_new = tanh(A0[t] + h0_prev @ Wh0^T) ----
    {
      const unsigned short* ar = cur + (size_t)arow_m * 2048;  // h0_prev row
      f32x8 acc0 = {}; f32x8 acc1 = {};
      for (int k = 0; k < HID; k += 32) {
        bf16x16 a = load_frag(ar, k + hsel * 8);
        acc0 = wmma_bf16(a, load_frag(wr0a, k + hsel * 8), acc0);
        acc1 = wmma_bf16(a, load_frag(wr0b, k + hsel * 8), acc1);
      }
      const float* a0t = A0 + (size_t)t * BATCH * HID;
#pragma unroll
      for (int j = 0; j < 8; ++j) {
        int m = m0 + hsel * 8 + j;
        float v0 = tanhf(acc0[j] + a0t[(size_t)m * HID + n_a]);
        float v1 = tanhf(acc1[j] + a0t[(size_t)m * HID + n_b]);
        nxt[(size_t)m * 2048 + n_a] = f32_to_bf16(v0);
        nxt[(size_t)m * 2048 + n_b] = f32_to_bf16(v1);
        if (t == T_STEPS - 1) {
          hfinal[(size_t)m * HID + n_a] = v0;
          hfinal[(size_t)m * HID + n_b] = v1;
        }
      }
    }
    grid_barrier(bar, gridDim.x);

    // ---- layer 1: h1_new = tanh(h0_new @ W_in1^T + h1_prev @ Wh1^T + b1) ----
    {
      const unsigned short* a1r = nxt + (size_t)arow_m * 2048;         // h0_new
      const unsigned short* a2r = cur + (size_t)arow_m * 2048 + 1024;  // h1_prev
      f32x8 acc0 = {}; f32x8 acc1 = {};
      for (int k = 0; k < HID; k += 32) {
        bf16x16 a = load_frag(a1r, k + hsel * 8);
        acc0 = wmma_bf16(a, load_frag(wr1a, k + hsel * 8), acc0);
        acc1 = wmma_bf16(a, load_frag(wr1b, k + hsel * 8), acc1);
      }
      for (int k = 0; k < HID; k += 32) {
        bf16x16 a = load_frag(a2r, k + hsel * 8);
        acc0 = wmma_bf16(a, load_frag(wr2a, k + hsel * 8), acc0);
        acc1 = wmma_bf16(a, load_frag(wr2b, k + hsel * 8), acc1);
      }
      unsigned short* h1out = H1all + (size_t)t * BATCH * HID;
#pragma unroll
      for (int j = 0; j < 8; ++j) {
        int m = m0 + hsel * 8 + j;
        float v0 = tanhf(acc0[j] + b1a);
        float v1 = tanhf(acc1[j] + b1b);
        unsigned short q0 = f32_to_bf16(v0), q1 = f32_to_bf16(v1);
        nxt[(size_t)m * 2048 + 1024 + n_a] = q0;
        nxt[(size_t)m * 2048 + 1024 + n_b] = q1;
        h1out[(size_t)m * HID + n_a] = q0;
        h1out[(size_t)m * HID + n_b] = q1;
        if (t == T_STEPS - 1) {
          hfinal[(size_t)(BATCH * HID) + (size_t)m * HID + n_a] = v0;
          hfinal[(size_t)(BATCH * HID) + (size_t)m * HID + n_b] = v1;
        }
      }
    }
    grid_barrier(bar, gridDim.x);
    unsigned short* tmp = cur; cur = nxt; nxt = tmp;
  }
}

// ---------------------------------------------------------------------------
// GEMM 2 (decoder): logits = H1all @ Wdec^T + bdec.
// M=8192 N=10240(pad) K=1024 — bulk of the FLOPs, fully parallel.
// LDS-staged: block tile 64(M) x 128(N) x 32(K), double-buffered. Each wave
// computes a 16x64 tile (4 accumulators, A-fragment reused 4x).
// Global traffic: 12KB per 262K MACs (~22 MACs/byte) vs 5.3 register-direct.
// ---------------------------------------------------------------------------
#define DEC_BM 64
#define DEC_BN 128
#define DEC_BK 32
#define DEC_B_OFF (DEC_BM * DEC_BK)                   // 2048 bf16 elements
#define DEC_STAGE_ELTS ((DEC_BM + DEC_BN) * DEC_BK)   // 6144 bf16 elements
#define DEC_KSTEPS (HID / DEC_BK)                     // 32

#if __has_builtin(__builtin_amdgcn_global_load_async_to_lds_b128) && \
    __has_builtin(__builtin_amdgcn_s_wait_asynccnt)
#define USE_ASYNC_LDS 1
// Builtin wants int-vector pointers: (global int4*, shared int4*, imm, imm)
typedef int v4i __attribute__((vector_size(16)));
typedef __attribute__((address_space(1))) v4i gv4i;
typedef __attribute__((address_space(3))) v4i lv4i;
#define ASYNC_CP128(gsrc, ldst) \
  __builtin_amdgcn_global_load_async_to_lds_b128((gv4i*)(gsrc), (lv4i*)(ldst), 0, 0)
#else
#define USE_ASYNC_LDS 0
#endif

__device__ __forceinline__ void dec_compute(const unsigned short* sm,
                                            int m_tile, int n_half,
                                            int hsel, int l15, f32x8* acc) {
  const unsigned short* ap = sm + (m_tile * 16 + l15) * DEC_BK + hsel * 8;
  FragU fa;
  fa.u[0] = *(const uint4*)(ap);
  fa.u[1] = *(const uint4*)(ap + 16);
#pragma unroll
  for (int sub = 0; sub < 4; ++sub) {
    const unsigned short* bp =
        sm + DEC_B_OFF + (n_half * 64 + sub * 16 + l15) * DEC_BK + hsel * 8;
    FragU fb;
    fb.u[0] = *(const uint4*)(bp);
    fb.u[1] = *(const uint4*)(bp + 16);
    acc[sub] = wmma_bf16(fa.v, fb.v, acc[sub]);
  }
}

__global__ void __launch_bounds__(256) gemm_dec_kernel(
    const unsigned short* __restrict H1all,   // [MROWS, HID] bf16
    const unsigned short* __restrict wdecb,   // [VOCP, HID] bf16
    const float* __restrict bdec,
    float* __restrict logits) {               // [MROWS, VOC]
  __shared__ __align__(16) unsigned short smem[2][DEC_STAGE_ELTS];

  const int tid  = threadIdx.x;
  const int lane = tid & 31;
  const int wave = tid >> 5;
  const int hsel = lane >> 4;
  const int l15  = lane & 15;
  const int m_tile = wave & 3;
  const int n_half = wave >> 2;
  const int m_blk = blockIdx.x * DEC_BM;
  const int n_blk = blockIdx.y * DEC_BN;

  // Per-thread staging assignment:
  //  A tile 64x32: 256 chunks of 8 bf16  -> 1 chunk/thread
  //  B tile 128x32: 512 chunks of 8 bf16 -> 2 chunks/thread (contiguous 16)
  const unsigned short* aS = H1all + (size_t)(m_blk + (tid >> 2)) * HID + (tid & 3) * 8;
  const unsigned short* bS = wdecb + (size_t)(n_blk + (tid >> 1)) * HID + (tid & 1) * 16;
  const int aD = (tid >> 2) * DEC_BK + (tid & 3) * 8;
  const int bD = DEC_B_OFF + (tid >> 1) * DEC_BK + (tid & 1) * 16;

  f32x8 acc[4] = {};

#if USE_ASYNC_LDS
  // Async global->LDS copies tracked with ASYNCcnt (3 per thread per stage).
  {
    unsigned short* d = &smem[0][0];
    ASYNC_CP128(aS, d + aD);
    ASYNC_CP128(bS, d + bD);
    ASYNC_CP128(bS + 8, d + bD + 8);
  }
  for (int s = 0; s < DEC_KSTEPS - 1; ++s) {
    unsigned short* d = &smem[(s + 1) & 1][0];
    const int ko = (s + 1) * DEC_BK;
    ASYNC_CP128(aS + ko, d + aD);
    ASYNC_CP128(bS + ko, d + bD);
    ASYNC_CP128(bS + ko + 8, d + bD + 8);
    __builtin_amdgcn_s_wait_asynccnt(3);   // stage s landed; stage s+1 in flight
    __syncthreads();
    dec_compute(&smem[s & 1][0], m_tile, n_half, hsel, l15, acc);
    __syncthreads();                        // protect buf (s+1)&1 reuse
  }
  __builtin_amdgcn_s_wait_asynccnt(0);
  __syncthreads();
  dec_compute(&smem[(DEC_KSTEPS - 1) & 1][0], m_tile, n_half, hsel, l15, acc);
#else
  // Fallback: regs -> ds_store double buffer.
  {
    uint4 ra  = *(const uint4*)(aS);
    uint4 rb0 = *(const uint4*)(bS);
    uint4 rb1 = *(const uint4*)(bS + 8);
    unsigned short* d = &smem[0][0];
    *(uint4*)(d + aD) = ra;
    *(uint4*)(d + bD) = rb0;
    *(uint4*)(d + bD + 8) = rb1;
  }
  __syncthreads();
  for (int s = 0; s < DEC_KSTEPS; ++s) {
    uint4 na = {}, nb0 = {}, nb1 = {};
    if (s + 1 < DEC_KSTEPS) {
      const int ko = (s + 1) * DEC_BK;
      na  = *(const uint4*)(aS + ko);
      nb0 = *(const uint4*)(bS + ko);
      nb1 = *(const uint4*)(bS + ko + 8);
    }
    dec_compute(&smem[s & 1][0], m_tile, n_half, hsel, l15, acc);
    __syncthreads();
    if (s + 1 < DEC_KSTEPS) {
      unsigned short* d = &smem[(s + 1) & 1][0];
      *(uint4*)(d + aD) = na;
      *(uint4*)(d + bD) = nb0;
      *(uint4*)(d + bD + 8) = nb1;
    }
    __syncthreads();
  }
#endif

  // Epilogue
#pragma unroll
  for (int sub = 0; sub < 4; ++sub) {
    const int n = n_blk + n_half * 64 + sub * 16 + l15;
    if (n < VOC) {
      const float bb = bdec[n];
#pragma unroll
      for (int j = 0; j < 8; ++j) {
        int m = m_blk + m_tile * 16 + hsel * 8 + j;
        logits[(size_t)m * VOC + n] = acc[sub][j] + bb;
      }
    }
  }
}

// ---------------------------------------------------------------------------
// Host launcher
// ---------------------------------------------------------------------------
extern "C" void kernel_launch(void* const* d_in, const int* in_sizes, int n_in,
                              void* d_out, int out_size, void* d_ws, size_t ws_size,
                              hipStream_t stream) {
  const float* emb    = (const float*)d_in[0];   // [V, E]
  const float* W_in0  = (const float*)d_in[1];   // [H, E]
  const float* Wh0    = (const float*)d_in[2];   // [H, H]
  const float* b0     = (const float*)d_in[3];   // [H]
  const float* W_in1  = (const float*)d_in[4];   // [H, H]
  const float* Wh1    = (const float*)d_in[5];   // [H, H]
  const float* b1     = (const float*)d_in[6];   // [H]
  const float* Wdec   = (const float*)d_in[7];   // [V, H]
  const float* bdec   = (const float*)d_in[8];   // [V]
  const float* hidden = (const float*)d_in[9];   // [2, B, H]
  const int*   tokens = (const int*)d_in[10];    // [T, B]
  float* out = (float*)d_out;                    // logits [T,B,V] ++ h_final [2,B,H]

  // Workspace carve-up (~101 MB)
  char* wsp = (char*)d_ws;
  size_t off = 0;
  auto wsalloc = [&](size_t bytes) -> void* {
    void* p = wsp + off;
    off += (bytes + 255) & ~(size_t)255;
    return p;
  };
  unsigned short* embb  = (unsigned short*)wsalloc((size_t)VOC * EMB * 2);
  unsigned short* w0b   = (unsigned short*)wsalloc((size_t)HID * EMB * 2);
  unsigned short* wh0b  = (unsigned short*)wsalloc((size_t)HID * HID * 2);
  unsigned short* w1b   = (unsigned short*)wsalloc((size_t)HID * HID * 2);
  unsigned short* wh1b  = (unsigned short*)wsalloc((size_t)HID * HID * 2);
  unsigned short* wdecb = (unsigned short*)wsalloc((size_t)VOCP * HID * 2);
  float*          A0    = (float*)wsalloc((size_t)MROWS * HID * 4);
  unsigned short* H1all = (unsigned short*)wsalloc((size_t)MROWS * HID * 2);
  unsigned short* hstA  = (unsigned short*)wsalloc((size_t)BATCH * 2048 * 2);
  unsigned short* hstB  = (unsigned short*)wsalloc((size_t)BATCH * 2048 * 2);
  GridBar*        bar   = (GridBar*)wsalloc(256);

  // 1) bf16 conversions (parallel, one-shot)
  cvt_f32_bf16_kernel<<<dim3(2048), 256, 0, stream>>>(emb, embb, VOC * EMB);
  cvt_f32_bf16_kernel<<<dim3(512), 256, 0, stream>>>(W_in0, w0b, HID * EMB);
  cvt_f32_bf16_kernel<<<dim3(512), 256, 0, stream>>>(Wh0, wh0b, HID * HID);
  cvt_f32_bf16_kernel<<<dim3(512), 256, 0, stream>>>(W_in1, w1b, HID * HID);
  cvt_f32_bf16_kernel<<<dim3(512), 256, 0, stream>>>(Wh1, wh1b, HID * HID);
  cvt_wdec_pad_kernel<<<dim3(2048), 256, 0, stream>>>(Wdec, wdecb);
  init_state_kernel<<<dim3(512), 256, 0, stream>>>(hidden, hstA, hstB, bar);

  // 2) input projection for all timesteps (hoisted out of the recurrence)
  gemm_a0_kernel<<<dim3(MROWS / 64, HID / 64), 256, 0, stream>>>(
      embb, w0b, b0, tokens, A0);

  // 3) serial recurrence: 16 persistent blocks, device barrier per phase
  rnn_recurrent_kernel<<<dim3(16), 256, 0, stream>>>(
      A0, wh0b, w1b, wh1b, b1, hstA, hstB, H1all,
      out + (size_t)MROWS * VOC, bar);

  // 4) decoder GEMM over all timesteps (LDS-staged, async if available)
  gemm_dec_kernel<<<dim3(MROWS / DEC_BM, VOCP / DEC_BN), 256, 0, stream>>>(
      H1all, wdecb, bdec, out);
}
